// AsymmetricAttention_70574902608343
// MI455X (gfx1250) — matrix operand order repaired and checked by reading, not compiled, not run on hardware
//
#include <hip/hip_runtime.h>

// ---------------- problem constants ----------------
#define NX 2048
#define NY 256
#define DX 3072
#define DY 1536
#define NH 24
#define HD 128
#define D3 9216           // 3*DX
#define CROP 192
#define SEQ 2240          // NX + CROP
#define DM  3072          // NH*HD joint row stride
#define EPS 1e-6f

typedef __attribute__((ext_vector_type(16))) _Float16 v16h;
typedef __attribute__((ext_vector_type(8)))  _Float16 v8h;
typedef __attribute__((ext_vector_type(8)))  float    v8f;
typedef __attribute__((ext_vector_type(4)))  unsigned int u32x4;
typedef __attribute__((ext_vector_type(8)))  unsigned int u32x8;

union AFrag { v16h v; v8h h[2]; };

__device__ __forceinline__ v8f wmma_f16(v16h a, v16h b, v8f c) {
  // D = A(16x32 f16) * B(32x16 f16) + C(16x16 f32)
  return __builtin_amdgcn_wmma_f32_16x16x32_f16(false, a, false, b, (short)0, c, false, false);
}

// ---- Tensor Data Mover: async load of a [32 keys x 128 dims] f16 tile to LDS ----
// D# per CDNA5 ISA: group0 {count,lds_addr,global_addr,type=2},
// group1 {data_size=2B, tensor dims DMxSEQ, tile 128x32, dim0 stride DM}.
__device__ __forceinline__ void tdm_load_k_tile(const _Float16* gsrc, _Float16* ldst) {
  u32x8 g1;
  g1[0] = 0x00010000u;                                   // workgroup_mask=0 | data_size=1 (2B)
  g1[1] = ((unsigned)DM & 0xFFFFu) << 16;                // tensor_dim0 lo16 (bits 63:48)
  g1[2] = (((unsigned)DM >> 16) & 0xFFFFu)
        | (((unsigned)SEQ & 0xFFFFu) << 16);             // tensor_dim0 hi16 | tensor_dim1 lo16
  g1[3] = (((unsigned)SEQ >> 16) & 0xFFFFu)
        | (128u << 16);                                  // tensor_dim1 hi16 | tile_dim0=128
  g1[4] = 32u;                                           // tile_dim1=32 | tile_dim2=0
  g1[5] = (unsigned)DM;                                  // tensor_dim0_stride lo32
  g1[6] = 0u;
  g1[7] = 0u;
  unsigned long long ga = (unsigned long long)(uintptr_t)gsrc;
  u32x4 g0;
  g0[0] = 1u;                                            // count=1, user mode
  g0[1] = (unsigned)(uintptr_t)ldst;                     // LDS byte offset (low 32 of generic ptr)
  g0[2] = (unsigned)ga;                                  // global_addr[31:0]
  g0[3] = (unsigned)(ga >> 32) | 0x80000000u;            // global_addr[56:32] | type=2 (bits 127:126)
  asm volatile("tensor_load_to_lds %0, %1" :: "s"(g0), "s"(g1) : "memory");
}

// ---------------- elementwise: f32 -> f16 cast (weights) ----------------
__global__ void cast_f16_kernel(const float* __restrict__ in, _Float16* __restrict__ out, long n) {
  long i = (long)blockIdx.x * blockDim.x + threadIdx.x;
  long stride = (long)gridDim.x * blockDim.x;
  for (; i < n; i += stride) out[i] = (_Float16)in[i];
}

// ---------------- RMS norm * (1 + scale), cast to f16 ----------------
__global__ void rms_mod_cast_kernel(const float* __restrict__ X, const float* __restrict__ sc,
                                    _Float16* __restrict__ out, int D) {
  int row = blockIdx.x;
  const float* xr = X + (size_t)row * D;
  __shared__ float red[256];
  float ss = 0.f;
  for (int i = threadIdx.x; i < D; i += 256) { float v = xr[i]; ss += v * v; }
  red[threadIdx.x] = ss;
  __syncthreads();
  for (int s = 128; s > 0; s >>= 1) {
    if ((int)threadIdx.x < s) red[threadIdx.x] += red[threadIdx.x + s];
    __syncthreads();
  }
  float rinv = rsqrtf(red[0] / (float)D + EPS);
  for (int i = threadIdx.x; i < D; i += 256)
    out[(size_t)row * D + i] = (_Float16)(xr[i] * rinv * (1.f + sc[i]));
}

// ---------------- WMMA GEMM: C[M,N] = A[M,K](f16) * W[N,K](f16)^T + bias ----------------
// block = 128 threads (4 waves); wave tile = (16*MF) x 64; block tile = (64*MF) x 64.
// MF row-fragments per wave: B frag reused MF times, A frags reused 4x.
template <int MF>
__global__ void __launch_bounds__(128) wmma_gemm_bias_kernel(
    const _Float16* __restrict__ A, int lda,
    const _Float16* __restrict__ W, int ldw,
    const float* __restrict__ bias,
    float* __restrict__ C, int ldc, int K) {
  const int lane = threadIdx.x & 31;
  const int wave = threadIdx.x >> 5;
  const int hi = lane >> 4;      // half-wave select
  const int ln = lane & 15;
  const int m0 = blockIdx.y * (64 * MF) + wave * (16 * MF);
  const int n0 = blockIdx.x * 64;

  v8f acc[MF][4] = {};
  const _Float16* arow[MF];
#pragma unroll
  for (int r = 0; r < MF; ++r) arow[r] = A + (size_t)(m0 + r * 16 + ln) * lda;

  for (int k0 = 0; k0 < K; k0 += 32) {
    AFrag a[MF];
#pragma unroll
    for (int r = 0; r < MF; ++r) {
      // A 16x32 layout: lanes 0-15: K 0..15; lanes 16-31: K 16..31
      a[r].v = *(const v16h*)(arow[r] + k0 + hi * 16);
    }
#pragma unroll
    for (int c = 0; c < 4; ++c) {
      // B 32x16 layout: lane = column; lanes 0-15: K 0..15; lanes 16-31: K 16..31
      v16h b = *(const v16h*)(W + (size_t)(n0 + c * 16 + ln) * ldw + k0 + hi * 16);
#pragma unroll
      for (int r = 0; r < MF; ++r) acc[r][c] = wmma_f16(a[r].v, b, acc[r][c]);
    }
  }
#pragma unroll
  for (int r = 0; r < MF; ++r) {
#pragma unroll
    for (int c = 0; c < 4; ++c) {
      int n = n0 + c * 16 + ln;
      float bv = bias ? bias[n] : 0.f;
#pragma unroll
      for (int j = 0; j < 8; ++j) {
        int m = m0 + r * 16 + j + hi * 8;  // C frag: vgpr j holds rows j / j+8
        C[(size_t)m * ldc + n] = acc[r][c][j] + bv;
      }
    }
  }
}

// ---------------- per-head RMS + RoPE for x branch; pack into joint q/k/v ----------------
__global__ void qkv_post_x_kernel(const float* __restrict__ qkv,
                                  const float* __restrict__ cosb, const float* __restrict__ sinb,
                                  const float* __restrict__ qw, const float* __restrict__ kw,
                                  _Float16* __restrict__ qh, _Float16* __restrict__ kh,
                                  _Float16* __restrict__ vh) {
  int s = blockIdx.x, h = blockIdx.y, t = threadIdx.x;
  size_t base = (size_t)s * D3 + (size_t)h * HD;
  float q0 = qkv[base + 2 * t],            q1 = qkv[base + 2 * t + 1];
  float k0 = qkv[base + DX + 2 * t],       k1 = qkv[base + DX + 2 * t + 1];
  float v0 = qkv[base + 2 * DX + 2 * t],   v1 = qkv[base + 2 * DX + 2 * t + 1];
  __shared__ float rq[64], rk[64];
  rq[t] = q0 * q0 + q1 * q1;
  rk[t] = k0 * k0 + k1 * k1;
  __syncthreads();
  for (int st = 32; st > 0; st >>= 1) {
    if (t < st) { rq[t] += rq[t + st]; rk[t] += rk[t + st]; }
    __syncthreads();
  }
  float qi = rsqrtf(rq[0] / (float)HD + EPS);
  float ki = rsqrtf(rk[0] / (float)HD + EPS);
  size_t rb = ((size_t)s * NH + h) * (HD / 2) + t;
  float c = cosb[rb], sn = sinb[rb];
  float qe = q0 * qi * qw[2 * t], qo = q1 * qi * qw[2 * t + 1];
  float ke = k0 * ki * kw[2 * t], ko = k1 * ki * kw[2 * t + 1];
  size_t ob = (size_t)s * DM + (size_t)h * HD;
  qh[ob + 2 * t]     = (_Float16)(qe * c - qo * sn);
  qh[ob + 2 * t + 1] = (_Float16)(qe * sn + qo * c);
  kh[ob + 2 * t]     = (_Float16)(ke * c - ko * sn);
  kh[ob + 2 * t + 1] = (_Float16)(ke * sn + ko * c);
  vh[ob + 2 * t]     = (_Float16)v0;
  vh[ob + 2 * t + 1] = (_Float16)v1;
}

// ---------------- per-head RMS for y branch (rows 0..CROP-1), pack at offset NX ----------------
__global__ void qkv_post_y_kernel(const float* __restrict__ qkv,
                                  const float* __restrict__ qw, const float* __restrict__ kw,
                                  _Float16* __restrict__ qh, _Float16* __restrict__ kh,
                                  _Float16* __restrict__ vh) {
  int s = blockIdx.x, h = blockIdx.y, t = threadIdx.x;
  size_t base = (size_t)s * D3 + (size_t)h * HD;
  float q0 = qkv[base + 2 * t],          q1 = qkv[base + 2 * t + 1];
  float k0 = qkv[base + DX + 2 * t],     k1 = qkv[base + DX + 2 * t + 1];
  float v0 = qkv[base + 2 * DX + 2 * t], v1 = qkv[base + 2 * DX + 2 * t + 1];
  __shared__ float rq[64], rk[64];
  rq[t] = q0 * q0 + q1 * q1;
  rk[t] = k0 * k0 + k1 * k1;
  __syncthreads();
  for (int st = 32; st > 0; st >>= 1) {
    if (t < st) { rq[t] += rq[t + st]; rk[t] += rk[t + st]; }
    __syncthreads();
  }
  float qi = rsqrtf(rq[0] / (float)HD + EPS);
  float ki = rsqrtf(rk[0] / (float)HD + EPS);
  size_t ob = (size_t)(NX + s) * DM + (size_t)h * HD;
  qh[ob + 2 * t]     = (_Float16)(q0 * qi * qw[2 * t]);
  qh[ob + 2 * t + 1] = (_Float16)(q1 * qi * qw[2 * t + 1]);
  kh[ob + 2 * t]     = (_Float16)(k0 * ki * kw[2 * t]);
  kh[ob + 2 * t + 1] = (_Float16)(k1 * ki * kw[2 * t + 1]);
  vh[ob + 2 * t]     = (_Float16)v0;
  vh[ob + 2 * t + 1] = (_Float16)v1;
}

// ---------------- flash attention: one head x 64 queries per block ----------------
// 4 waves x 16 query rows. K-tile = 32 keys. K staged by TDM (double-buffered,
// issued by wave 0, overlapped with compute); V staged transposed by all waves.
__global__ void __launch_bounds__(128) flash_attn_kernel(
    const _Float16* __restrict__ Qh, const _Float16* __restrict__ Kh,
    const _Float16* __restrict__ Vh, _Float16* __restrict__ Oh) {
  const int h = blockIdx.y;
  const int qbase = blockIdx.x * 64;
  const int lane = threadIdx.x & 31;
  const int wave = threadIdx.x >> 5;
  const int hi = lane >> 4;
  const int ln = lane & 15;

  __shared__ __align__(32) _Float16 ldsK[2][32 * 128];   // [key][dim], TDM destination
  __shared__ __align__(32) _Float16 ldsVT[2][128 * 32];  // [dim][key]
  __shared__ __align__(32) _Float16 ldsP[4][16 * 32];    // per-wave P tile

  // Q fragments (16 rows x 128 dims = 4 A-frags), kept in VGPRs
  AFrag qa[4];
  {
    const _Float16* qrow = Qh + (size_t)(qbase + wave * 16 + ln) * DM + h * HD;
#pragma unroll
    for (int ks = 0; ks < 4; ++ks) {
      qa[ks].v = *(const v16h*)(qrow + ks * 32 + hi * 16);
    }
  }

  v8f o[8] = {};
  float mrow[8], lrow[8];
#pragma unroll
  for (int j = 0; j < 8; ++j) { mrow[j] = -__builtin_inff(); lrow[j] = 0.f; }
  const float scale = 0.08838834764831845f;  // 1/sqrt(128)

  const int T = SEQ / 32;  // 70 key tiles

  // manual transposed V staging into back buffer
  auto stage_v = [&](int t, int b) {
    int key = threadIdx.x >> 2;           // 0..31
    int seg = (threadIdx.x & 3) * 32;     // 0,32,64,96
    const _Float16* vrow = Vh + (size_t)(t * 32 + key) * DM + h * HD + seg;
    v16h vv0 = *(const v16h*)(vrow);
    v16h vv1 = *(const v16h*)(vrow + 16);
#pragma unroll
    for (int i = 0; i < 16; ++i) {
      ldsVT[b][(seg + i) * 32 + key]      = vv0[i];
      ldsVT[b][(seg + 16 + i) * 32 + key] = vv1[i];
    }
  };

  // prologue: stage tile 0
  if (wave == 0) tdm_load_k_tile(Kh + (size_t)h * HD, &ldsK[0][0]);
  stage_v(0, 0);

  for (int t = 0; t < T; ++t) {
    const int buf = t & 1;
    // ---- kick off next tile into the back buffer while we compute ----
    if (t + 1 < T) {
      if (wave == 0)
        tdm_load_k_tile(Kh + (size_t)(t + 1) * 32 * DM + h * HD, &ldsK[buf ^ 1][0]);
      stage_v(t + 1, buf ^ 1);
    }
    // ---- wait for tile t's TDM (<=1 outstanding leaves t+1 in flight) ----
    if (wave == 0) {
      if (t + 1 < T) __builtin_amdgcn_s_wait_tensorcnt((short)1);
      else           __builtin_amdgcn_s_wait_tensorcnt((short)0);
    }
    __syncthreads();

    // ---- scores: S(16q x 32k) = Q(16x128) * K_tile^T, two 16x16 C frags ----
    v8f sc[2] = {};
#pragma unroll
    for (int c = 0; c < 2; ++c) {
#pragma unroll
      for (int ks = 0; ks < 4; ++ks) {
        v16h b = *(const v16h*)(&ldsK[buf][(c * 16 + ln) * 128 + ks * 32 + hi * 16]);
        sc[c] = wmma_f16(qa[ks].v, b, sc[c]);
      }
    }

    // ---- online softmax; build P in LDS (C-layout -> A-layout round trip) ----
#pragma unroll
    for (int j = 0; j < 8; ++j) {
      float s0 = sc[0][j] * scale, s1 = sc[1][j] * scale;
      float rmax = fmaxf(s0, s1);
      rmax = fmaxf(rmax, __shfl_xor(rmax, 1, 32));
      rmax = fmaxf(rmax, __shfl_xor(rmax, 2, 32));
      rmax = fmaxf(rmax, __shfl_xor(rmax, 4, 32));
      rmax = fmaxf(rmax, __shfl_xor(rmax, 8, 32));
      float mnew = fmaxf(mrow[j], rmax);
      float alpha = __expf(mrow[j] - mnew);
      float p0 = __expf(s0 - mnew), p1 = __expf(s1 - mnew);
      float rs = p0 + p1;
      rs += __shfl_xor(rs, 1, 32);
      rs += __shfl_xor(rs, 2, 32);
      rs += __shfl_xor(rs, 4, 32);
      rs += __shfl_xor(rs, 8, 32);
      lrow[j] = lrow[j] * alpha + rs;
      mrow[j] = mnew;
#pragma unroll
      for (int f = 0; f < 8; ++f) o[f][j] *= alpha;
      int prow = j + hi * 8;
      ldsP[wave][prow * 32 + ln]      = (_Float16)p0;
      ldsP[wave][prow * 32 + 16 + ln] = (_Float16)p1;
    }
    __syncthreads();

    // ---- O(16x128) += P(16x32) * V_tile(32x128): 8 WMMAs ----
    AFrag pa;
    pa.v = *(const v16h*)(&ldsP[wave][ln * 32 + hi * 16]);
#pragma unroll
    for (int f = 0; f < 8; ++f) {
      v16h b = *(const v16h*)(&ldsVT[buf][(f * 16 + ln) * 32 + hi * 16]);
      o[f] = wmma_f16(pa.v, b, o[f]);
    }
    __syncthreads();  // all reads of buf done before it is restaged next iter
  }

  // ---- normalize and write joint output (f16 for projection GEMMs) ----
#pragma unroll
  for (int j = 0; j < 8; ++j) {
    float linv = 1.f / lrow[j];
    int row = qbase + wave * 16 + j + hi * 8;
#pragma unroll
    for (int f = 0; f < 8; ++f)
      Oh[(size_t)row * DM + h * HD + f * 16 + ln] = (_Float16)(o[f][j] * linv);
  }
}

// ---------------- fill zero-padded tail of y_out with bias ----------------
__global__ void yout_tail_kernel(float* __restrict__ yout, const float* __restrict__ bias) {
  int idx = blockIdx.x * 256 + threadIdx.x;
  if (idx < (NY - CROP) * DY) {
    int r = idx / DY, c = idx % DY;
    yout[(size_t)(CROP + r) * DY + c] = bias[c];
  }
}

// ================= host-side orchestration =================
extern "C" void kernel_launch(void* const* d_in, const int* in_sizes, int n_in,
                              void* d_out, int out_size, void* d_ws, size_t ws_size,
                              hipStream_t stream) {
  (void)in_sizes; (void)n_in; (void)out_size; (void)ws_size;
  const float* x        = (const float*)d_in[0];
  const float* y        = (const float*)d_in[1];
  const float* scale_x  = (const float*)d_in[2];
  const float* scale_y  = (const float*)d_in[3];
  const float* rope_cos = (const float*)d_in[4];
  const float* rope_sin = (const float*)d_in[5];
  const float* w_qkv_x  = (const float*)d_in[6];
  const float* b_qkv_x  = (const float*)d_in[7];
  const float* w_qkv_y  = (const float*)d_in[8];
  const float* b_qkv_y  = (const float*)d_in[9];
  const float* qnx_w    = (const float*)d_in[10];
  const float* knx_w    = (const float*)d_in[11];
  const float* qny_w    = (const float*)d_in[12];
  const float* kny_w    = (const float*)d_in[13];
  const float* w_proj_x = (const float*)d_in[14];
  const float* b_proj_x = (const float*)d_in[15];
  const float* w_proj_y = (const float*)d_in[16];
  const float* b_proj_y = (const float*)d_in[17];

  char* ws = (char*)d_ws;
  size_t off = 0;
  auto alloc = [&](size_t bytes) -> void* {
    off = (off + 255) & ~(size_t)255;
    void* p = ws + off;
    off += bytes;
    return p;
  };

  _Float16* xm_h  = (_Float16*)alloc((size_t)NX * DX * 2);
  _Float16* ym_h  = (_Float16*)alloc((size_t)NY * DY * 2);
  _Float16* wqx_h = (_Float16*)alloc((size_t)D3 * DX * 2);
  _Float16* wqy_h = (_Float16*)alloc((size_t)D3 * DY * 2);
  _Float16* wpx_h = (_Float16*)alloc((size_t)DX * DX * 2);
  _Float16* wpy_h = (_Float16*)alloc((size_t)DY * DX * 2);
  float*    qkvx  = (float*)alloc((size_t)NX * D3 * 4);
  float*    qkvy  = (float*)alloc((size_t)NY * D3 * 4);
  _Float16* q_h   = (_Float16*)alloc((size_t)SEQ * DM * 2);
  _Float16* k_h   = (_Float16*)alloc((size_t)SEQ * DM * 2);
  _Float16* v_h   = (_Float16*)alloc((size_t)SEQ * DM * 2);
  _Float16* xy_h  = (_Float16*)alloc((size_t)SEQ * DM * 2);

  // 1) one-time weight casts (f16 weights then live in the 192MB L2)
  cast_f16_kernel<<<2048, 256, 0, stream>>>(w_qkv_x, wqx_h, (long)D3 * DX);
  cast_f16_kernel<<<2048, 256, 0, stream>>>(w_qkv_y, wqy_h, (long)D3 * DY);
  cast_f16_kernel<<<2048, 256, 0, stream>>>(w_proj_x, wpx_h, (long)DX * DX);
  cast_f16_kernel<<<2048, 256, 0, stream>>>(w_proj_y, wpy_h, (long)DY * DX);

  // 2) RMS + (1+scale) modulation
  rms_mod_cast_kernel<<<NX, 256, 0, stream>>>(x, scale_x, xm_h, DX);
  rms_mod_cast_kernel<<<NY, 256, 0, stream>>>(y, scale_y, ym_h, DY);

  // 3) QKV GEMMs (WMMA f16 -> f32), 32x64 wave tiles
  wmma_gemm_bias_kernel<2><<<dim3(D3 / 64, NX / 128), 128, 0, stream>>>(
      xm_h, DX, wqx_h, DX, b_qkv_x, qkvx, D3, DX);
  wmma_gemm_bias_kernel<2><<<dim3(D3 / 64, NY / 128), 128, 0, stream>>>(
      ym_h, DY, wqy_h, DY, b_qkv_y, qkvy, D3, DY);

  // 4) per-head RMS (+RoPE for x); pack joint q/k/v in f16
  qkv_post_x_kernel<<<dim3(NX, NH), 64, 0, stream>>>(qkvx, rope_cos, rope_sin,
                                                     qnx_w, knx_w, q_h, k_h, v_h);
  qkv_post_y_kernel<<<dim3(CROP, NH), 64, 0, stream>>>(qkvy, qny_w, kny_w, q_h, k_h, v_h);

  // 5) flash attention over joint sequence (TDM-staged K tiles)
  flash_attn_kernel<<<dim3(SEQ / 64, NH), 128, 0, stream>>>(q_h, k_h, v_h, xy_h);

  // 6) output projections
  float* x_out = (float*)d_out;
  float* y_out = (float*)d_out + (size_t)NX * DX;
  wmma_gemm_bias_kernel<2><<<dim3(DX / 64, NX / 128), 128, 0, stream>>>(
      xy_h, DM, wpx_h, DX, b_proj_x, x_out, DX, DX);
  wmma_gemm_bias_kernel<3><<<dim3(DY / 64, 1), 128, 0, stream>>>(
      xy_h + (size_t)NX * DM, DM, wpy_h, DX, b_proj_y, y_out, DY, DX);
  yout_tail_kernel<<<((NY - CROP) * DY + 255) / 256, 256, 0, stream>>>(y_out, b_proj_y);
}
